// Transformer_84722524880901
// MI455X (gfx1250) — matrix operand order repaired and checked by reading, not compiled, not run on hardware
//
#include <hip/hip_runtime.h>
#include <math.h>

// ---------------------------------------------------------------------------
// ViT-style encoder (L=6, D=512, H=8, DH=64, MLP=2048, B=8, N=1024) on gfx1250.
// All GEMMs run on v_wmma_f32_16x16x32_bf16 (wave32 WMMA). f32 for LN/softmax
// statistics, residuals and the attn_full output. Weights are transposed +
// converted to bf16 [N,K] once per launch so A and B WMMA fragments are
// contiguous 16B/32B global loads matching the CDNA5 VGPR layouts.
// Wave tile 64x64 (4x4 WMMA accumulators) -> 1.0 b128 loads per WMMA.
// ---------------------------------------------------------------------------

typedef __bf16 bf16;
typedef __attribute__((ext_vector_type(16))) __bf16 v16bf;
typedef __attribute__((ext_vector_type(8)))  __bf16 v8bf;
typedef __attribute__((ext_vector_type(8)))  float  v8f;

#define LAYERS 6
#define DIM    512
#define HEADS  8
#define DHEAD  64
#define MLPD   2048
#define BATCH  8
#define SEQ    1024
#define BNROWS (BATCH*SEQ)

__device__ __forceinline__ float wave_sum(float v) {
#pragma unroll
  for (int off = 16; off > 0; off >>= 1) v += __shfl_xor(v, off, 32);
  return v;
}
__device__ __forceinline__ float wave_max(float v) {
#pragma unroll
  for (int off = 16; off > 0; off >>= 1) v = fmaxf(v, __shfl_xor(v, off, 32));
  return v;
}

// ---------------------------------------------------------------------------
// Generic bf16 WMMA GEMM:  C[z][M,N] (+bias +residual, opt GELU, opt bf16 out)
//   A : bf16 [M,K] row-major (per-batch stride sA)
//   Bt: bf16 [N,K] row-major, i.e. B transposed (per-batch stride sB)
// Block: 256 threads = 8 waves; 256x128 block tile; wave tile 64x64
// (4x4 WMMA accumulators = 128 acc VGPRs).
// ---------------------------------------------------------------------------
__global__ __launch_bounds__(256)
void gemm_wmma_bf16(const bf16* __restrict__ A,  long lda, long sA,
                    const bf16* __restrict__ Bt, long ldb, long sB,
                    void* __restrict__ Cv, long ldc, long sC, int c_bf16,
                    const float* __restrict__ bias,
                    const float* __restrict__ Res, long ldr, long sR,
                    int M, int N, int K, int gelu)
{
  const int lane = threadIdx.x & 31;
  const int wid  = threadIdx.x >> 5;
  const int wm   = wid & 3;        // 4 waves along M
  const int wn   = wid >> 2;       // 2 waves along N
  const int half = lane >> 4;      // lane group 0/1
  const int l16  = lane & 15;
  const long z   = blockIdx.z;

  A  += z * sA;
  Bt += z * sB;
  if (Res) Res += z * sR;
  const long cOff = z * sC;

  const int m0 = blockIdx.y * 256 + wm * 64;
  const int n0 = blockIdx.x * 128 + wn * 64;

  v8f acc[4][4];
#pragma unroll
  for (int t = 0; t < 4; t++)
#pragma unroll
    for (int u = 0; u < 4; u++)
#pragma unroll
      for (int r = 0; r < 8; r++) acc[t][u][r] = 0.0f;

  for (int kk = 0; kk < K; kk += 32) {
    // A fragments: 16x32 bf16 tile per 16 rows. lane m=l16; elems 0..7 hold
    // K = kk + half*8 + e, elems 8..15 hold K = kk + 16 + half*8 + (e-8).
    v16bf afrag[4];
#pragma unroll
    for (int t = 0; t < 4; t++) {
      int row = m0 + t * 16 + l16;
      row = row < M ? row : M - 1;
      const bf16* p = A + (long)row * lda + kk + half * 8;
      v8bf lo = *(const v8bf*)p;
      v8bf hi = *(const v8bf*)(p + 16);
#pragma unroll
      for (int i = 0; i < 8; i++) { afrag[t][i] = lo[i]; afrag[t][8 + i] = hi[i]; }
      if (t == 0)  // speculative prefetch of next K-step (OOB is silently dropped)
        __builtin_prefetch((const void*)(p + 32), 0, 1);
    }
    // B fragments from Bt[N,K]: lane col=n0+u*16+l16; same K order as A.
    v16bf bfrag[4];
#pragma unroll
    for (int u = 0; u < 4; u++) {
      int col = n0 + u * 16 + l16;
      col = col < N ? col : N - 1;
      const bf16* p = Bt + (long)col * ldb + kk + half * 8;
      v8bf lo = *(const v8bf*)p;
      v8bf hi = *(const v8bf*)(p + 16);
#pragma unroll
      for (int i = 0; i < 8; i++) { bfrag[u][i] = lo[i]; bfrag[u][8 + i] = hi[i]; }
      if (u == 0)
        __builtin_prefetch((const void*)(p + 32), 0, 1);
    }
#pragma unroll
    for (int t = 0; t < 4; t++)
#pragma unroll
      for (int u = 0; u < 4; u++)
        acc[t][u] = __builtin_amdgcn_wmma_f32_16x16x32_bf16(
            false, afrag[t], false, bfrag[u], (short)0, acc[t][u], false, false);
  }

  // Epilogue. C/D layout: m = tile + half*8 + r, n = tile + l16.
#pragma unroll
  for (int t = 0; t < 4; t++) {
#pragma unroll
    for (int u = 0; u < 4; u++) {
      int n = n0 + u * 16 + l16;
      if (n >= N) continue;
      float bv = bias ? bias[n] : 0.0f;
#pragma unroll
      for (int r = 0; r < 8; r++) {
        int m = m0 + t * 16 + half * 8 + r;
        if (m >= M) continue;
        float v = acc[t][u][r] + bv;
        if (Res) v += Res[(long)m * ldr + n];
        if (gelu) v = 0.5f * v * (1.0f + erff(v * 0.70710678118654752f));
        if (c_bf16) ((bf16*)Cv)[cOff + (long)m * ldc + n] = (bf16)v;
        else        ((float*)Cv)[cOff + (long)m * ldc + n] = v;
      }
    }
  }
}

// ---------------------------------------------------------------------------
// LayerNorm(D=512) -> bf16. One wave per row, 16 elems/lane, wave32 reductions.
// ---------------------------------------------------------------------------
__global__ __launch_bounds__(256)
void ln_to_bf16(const float* __restrict__ X, const float* __restrict__ g,
                const float* __restrict__ b, bf16* __restrict__ out)
{
  const int lane = threadIdx.x & 31;
  const long row = (long)blockIdx.x * 8 + (threadIdx.x >> 5);
  const float* xr = X + row * DIM;
  float v[16];
  float s = 0.f, sq = 0.f;
#pragma unroll
  for (int i = 0; i < 16; i++) {
    float t = xr[lane + i * 32];
    v[i] = t; s += t; sq += t * t;
  }
  s = wave_sum(s); sq = wave_sum(sq);
  float mean = s * (1.0f / DIM);
  float var  = fmaxf(sq * (1.0f / DIM) - mean * mean, 0.0f);
  float rs   = rsqrtf(var + 1e-5f);
  bf16* orow = out + row * DIM;
#pragma unroll
  for (int i = 0; i < 16; i++) {
    int c = lane + i * 32;
    orow[c] = (bf16)((v[i] - mean) * rs * g[c] + b[c]);
  }
}

// ---------------------------------------------------------------------------
// Row softmax over 1024 elems. One wave per row, 32 elems/lane.
// outF != nullptr -> f32 output (attn_full); else bf16 output (per-head attn).
// ---------------------------------------------------------------------------
__global__ __launch_bounds__(256)
void softmax1024(const float* __restrict__ S, float* __restrict__ outF,
                 bf16* __restrict__ outB)
{
  const int lane = threadIdx.x & 31;
  const long row = (long)blockIdx.x * 8 + (threadIdx.x >> 5);
  const float* r = S + row * SEQ;
  float v[32];
  float mx = -3.4e38f;
#pragma unroll
  for (int i = 0; i < 32; i++) { v[i] = r[lane + i * 32]; mx = fmaxf(mx, v[i]); }
  mx = wave_max(mx);
  float s = 0.f;
#pragma unroll
  for (int i = 0; i < 32; i++) { v[i] = __expf(v[i] - mx); s += v[i]; }
  s = wave_sum(s);
  float inv = 1.0f / s;
  if (outF) {
    float* o = outF + row * SEQ;
#pragma unroll
    for (int i = 0; i < 32; i++) o[lane + i * 32] = v[i] * inv;
  } else {
    bf16* o = outB + row * SEQ;
#pragma unroll
    for (int i = 0; i < 32; i++) o[lane + i * 32] = (bf16)(v[i] * inv);
  }
}

// W[K,N] f32 -> Wt[N,K] bf16 (coalesced writes; one-time per launch).
__global__ void wconv_t(const float* __restrict__ W, bf16* __restrict__ Wt,
                        int K, int N)
{
  long idx = (long)blockIdx.x * blockDim.x + threadIdx.x;
  if (idx >= (long)K * N) return;
  long n = idx / K, k = idx % K;
  Wt[idx] = (bf16)W[k * (long)N + n];
}

// qkv[BN,1536] f32 -> q_full/k_full [BN,512] bf16 (scale_full folded into q),
// q_h/k_h [B,H,N,DH] bf16 (per-head scale folded into q), v_t [B,H,DH,N] bf16.
__global__ void qkv_split(const float* __restrict__ qkv,
                          bf16* __restrict__ q_full, bf16* __restrict__ k_full,
                          bf16* __restrict__ q_h, bf16* __restrict__ k_h,
                          bf16* __restrict__ v_t)
{
  const float scale_full = 0.044194173824159216f;  // 512^-0.5
  const float scale_head = 0.125f;                 // 64^-0.5
  long idx = (long)blockIdx.x * blockDim.x + threadIdx.x;
  if (idx >= (long)BNROWS * DIM) return;
  long row = idx >> 9;          // /512
  int  c   = (int)(idx & 511);
  long b = row >> 10; long n = row & 1023;
  int  h = c >> 6;    int  dh = c & 63;
  const float* base = qkv + row * (3 * DIM);
  float q = base[c], k = base[DIM + c], v = base[2 * DIM + c];
  q_full[idx] = (bf16)(q * scale_full);
  k_full[idx] = (bf16)k;
  long hb = (((b * HEADS + h) * SEQ) + n) * DHEAD + dh;
  q_h[hb] = (bf16)(q * scale_head);
  k_h[hb] = (bf16)k;
  v_t[(((b * HEADS + h) * DHEAD) + dh) * SEQ + n] = (bf16)v;
}

__global__ void copy_f32(const float* __restrict__ src, float* __restrict__ dst, long n)
{
  long idx = (long)blockIdx.x * blockDim.x + threadIdx.x;
  if (idx < n) dst[idx] = src[idx];
}

// ---------------------------------------------------------------------------
extern "C" void kernel_launch(void* const* d_in, const int* in_sizes, int n_in,
                              void* d_out, int out_size, void* d_ws, size_t ws_size,
                              hipStream_t stream)
{
  (void)in_sizes; (void)n_in; (void)out_size; (void)ws_size;
  const float* x_in  = (const float*)d_in[0];
  const float* ln1g  = (const float*)d_in[1];
  const float* ln1b  = (const float*)d_in[2];
  const float* wqkv  = (const float*)d_in[3];
  const float* wo    = (const float*)d_in[4];
  const float* bo    = (const float*)d_in[5];
  const float* ln2g  = (const float*)d_in[6];
  const float* ln2b  = (const float*)d_in[7];
  const float* w1    = (const float*)d_in[8];
  const float* b1    = (const float*)d_in[9];
  const float* w2    = (const float*)d_in[10];
  const float* b2    = (const float*)d_in[11];

  char* ws = (char*)d_ws;
  size_t off = 0;
  auto alloc = [&](size_t bytes) -> char* {
    char* p = ws + off;
    off = (off + bytes + 255) & ~(size_t)255;
    return p;
  };

  float* x_cur   = (float*)alloc((size_t)BNROWS * DIM * 4);
  bf16*  h_bf    = (bf16*) alloc((size_t)BNROWS * DIM * 2);
  float* qkv_f   = (float*)alloc((size_t)BNROWS * 3 * DIM * 4);
  bf16*  q_full  = (bf16*) alloc((size_t)BNROWS * DIM * 2);
  bf16*  k_full  = (bf16*) alloc((size_t)BNROWS * DIM * 2);
  bf16*  q_h     = (bf16*) alloc((size_t)BNROWS * DIM * 2);
  bf16*  k_h     = (bf16*) alloc((size_t)BNROWS * DIM * 2);
  bf16*  v_t     = (bf16*) alloc((size_t)BNROWS * DIM * 2);
  float* scores  = (float*)alloc((size_t)BATCH * SEQ * SEQ * 4);
  bf16*  attn_bf = (bf16*) alloc((size_t)BATCH * SEQ * SEQ * 2);
  bf16*  o_bf    = (bf16*) alloc((size_t)BNROWS * DIM * 2);
  bf16*  ff_bf   = (bf16*) alloc((size_t)BNROWS * MLPD * 2);
  bf16*  wqkv_t  = (bf16*) alloc((size_t)LAYERS * 3 * DIM * DIM * 2);
  bf16*  wo_t    = (bf16*) alloc((size_t)LAYERS * DIM * DIM * 2);
  bf16*  w1_t    = (bf16*) alloc((size_t)LAYERS * DIM * MLPD * 2);
  bf16*  w2_t    = (bf16*) alloc((size_t)LAYERS * MLPD * DIM * 2);

  float* out_x    = (float*)d_out;
  float* out_attn = out_x + (size_t)BNROWS * DIM;

  // ---- one-time (per launch) weight transpose+bf16 conversion ----
  for (int l = 0; l < LAYERS; l++) {
    long e;
    e = (long)DIM * 3 * DIM;
    wconv_t<<<(e + 255) / 256, 256, 0, stream>>>(
        wqkv + (long)l * e, wqkv_t + (long)l * e, DIM, 3 * DIM);
    e = (long)DIM * DIM;
    wconv_t<<<(e + 255) / 256, 256, 0, stream>>>(
        wo + (long)l * e, wo_t + (long)l * e, DIM, DIM);
    e = (long)DIM * MLPD;
    wconv_t<<<(e + 255) / 256, 256, 0, stream>>>(
        w1 + (long)l * e, w1_t + (long)l * e, DIM, MLPD);
    wconv_t<<<(e + 255) / 256, 256, 0, stream>>>(
        w2 + (long)l * e, w2_t + (long)l * e, MLPD, DIM);
  }
  {
    long e = (long)BNROWS * DIM;
    copy_f32<<<(e + 255) / 256, 256, 0, stream>>>(x_in, x_cur, e);
  }

  const dim3 blk(256);
  const long sBD  = (long)SEQ * DIM;     // per-batch stride, [N,D] slabs
  const long sNN  = (long)SEQ * SEQ;     // per-batch stride, [N,N] slabs
  const long sHND = (long)HEADS * SEQ * DHEAD;
  const long sHDN = (long)HEADS * DHEAD * SEQ;

  for (int l = 0; l < LAYERS; l++) {
    // --- LN1 ---
    ln_to_bf16<<<BNROWS / 8, blk, 0, stream>>>(x_cur, ln1g + l * DIM, ln1b + l * DIM, h_bf);

    // --- QKV: [8192,512] x [512,1536] -> f32 [8192,1536] ---
    gemm_wmma_bf16<<<dim3(12, 32, 1), blk, 0, stream>>>(
        h_bf, DIM, 0, wqkv_t + (long)l * 3 * DIM * DIM, DIM, 0,
        qkv_f, 3 * DIM, 0, 0, nullptr, nullptr, 0, 0,
        BNROWS, 3 * DIM, DIM, 0);

    // --- split + scale-fold + head/transpose layouts ---
    {
      long e = (long)BNROWS * DIM;
      qkv_split<<<(e + 255) / 256, blk, 0, stream>>>(qkv_f, q_full, k_full, q_h, k_h, v_t);
    }

    // --- full-dim attention matrix: S = (q*sf) K^T -> softmax -> d_out ---
    gemm_wmma_bf16<<<dim3(8, 4, BATCH), blk, 0, stream>>>(
        q_full, DIM, sBD, k_full, DIM, sBD,
        scores, SEQ, sNN, 0, nullptr, nullptr, 0, 0,
        SEQ, SEQ, DIM, 0);
    softmax1024<<<BATCH * SEQ / 8, blk, 0, stream>>>(
        scores, out_attn + (long)l * BATCH * sNN, nullptr);

    // --- per-head attention ---
    for (int h = 0; h < HEADS; h++) {
      // S_h = (q_h*s) k_h^T   [1024,1024], K=64
      gemm_wmma_bf16<<<dim3(8, 4, BATCH), blk, 0, stream>>>(
          q_h + (long)h * SEQ * DHEAD, DHEAD, sHND,
          k_h + (long)h * SEQ * DHEAD, DHEAD, sHND,
          scores, SEQ, sNN, 0, nullptr, nullptr, 0, 0,
          SEQ, SEQ, DHEAD, 0);
      softmax1024<<<BATCH * SEQ / 8, blk, 0, stream>>>(scores, nullptr, attn_bf);
      // O_h = attn x V  [1024,64], K=1024; write bf16 strided into o_bf[:, h*64:]
      gemm_wmma_bf16<<<dim3(1, 4, BATCH), blk, 0, stream>>>(
          attn_bf, SEQ, sNN,
          v_t + (long)h * DHEAD * SEQ, SEQ, sHDN,
          o_bf + h * DHEAD, DIM, sBD, 1, nullptr, nullptr, 0, 0,
          SEQ, DHEAD, SEQ, 0);
    }

    // --- output projection + bias + residual -> x_cur ---
    gemm_wmma_bf16<<<dim3(4, 32, 1), blk, 0, stream>>>(
        o_bf, DIM, 0, wo_t + (long)l * DIM * DIM, DIM, 0,
        x_cur, DIM, 0, 0, bo + l * DIM, x_cur, DIM, 0,
        BNROWS, DIM, DIM, 0);

    // --- LN2 ---
    ln_to_bf16<<<BNROWS / 8, blk, 0, stream>>>(x_cur, ln2g + l * DIM, ln2b + l * DIM, h_bf);

    // --- FFN1: GELU(h W1 + b1) -> bf16 [8192,2048] ---
    gemm_wmma_bf16<<<dim3(16, 32, 1), blk, 0, stream>>>(
        h_bf, DIM, 0, w1_t + (long)l * DIM * MLPD, DIM, 0,
        ff_bf, MLPD, 0, 1, b1 + l * MLPD, nullptr, 0, 0,
        BNROWS, MLPD, DIM, 1);

    // --- FFN2: ff W2 + b2 + residual -> x_cur ---
    gemm_wmma_bf16<<<dim3(4, 32, 1), blk, 0, stream>>>(
        ff_bf, MLPD, 0, w2_t + (long)l * MLPD * DIM, MLPD, 0,
        x_cur, DIM, 0, 0, b2 + l * DIM, x_cur, DIM, 0,
        BNROWS, DIM, MLPD, 0);
  }

  // final hidden states -> d_out[0 : B*N*D]
  {
    long e = (long)BNROWS * DIM;
    copy_f32<<<(e + 255) / 256, 256, 0, stream>>>(x_cur, out_x, e);
  }
}
